// DeepseekMoE_18038862643810
// MI455X (gfx1250) — compile-verified, matrix operand order checked by reading, not run on hardware
//
#include <hip/hip_runtime.h>

// ---------------------------------------------------------------------------
// DeepSeek MoE for MI455X (gfx1250): sparse grouped-GEMM with bf16 WMMA.
// - fp32 weights converted to bf16 in-register while staging into LDS
// - bf16 activation tiles moved memory->LDS with GLOBAL_LOAD_ASYNC_TO_LDS
// - double-buffered LDS, one barrier per K-step
// ---------------------------------------------------------------------------

typedef __attribute__((ext_vector_type(16))) __bf16       v16bf;
typedef __attribute__((ext_vector_type(8)))  float        v8f;
typedef __attribute__((ext_vector_type(4)))  unsigned int u32x4;

namespace {
constexpr int T_TOK  = 1024;
constexpr int H_DIM  = 2048;
constexpr int E_NUM  = 16;
constexpr int I_DIM  = 1408;   // routed intermediate
constexpr int IS_DIM = 2816;   // shared intermediate
constexpr int TOP_K  = 4;
constexpr float ROUTED_SCALE = 2.5f;

constexpr int BM  = 128;       // output tile rows
constexpr int BN  = 128;       // output tile cols
constexpr int BK  = 32;        // K step (one WMMA K)
constexpr int LDT = BK + 8;    // LDS row stride (bf16 elems): 16B-aligned, staggers banks

constexpr int CAP_ROWS  = 6144;            // 4*T plus per-expert 128-padding headroom
constexpr int ROW_TILES = CAP_ROWS / BM;   // 48
}

// ------------------------------ async helpers ------------------------------

// Two 16-byte memory->LDS async copies per lane (ASYNCcnt).  The immediate
// offset is added to BOTH the global and LDS addresses (ISA 10.x), which is
// exactly what a contiguous 32B-per-lane copy wants.
__device__ __forceinline__ void async_cp_b128x2(unsigned lds_addr, const void* gaddr) {
  asm volatile("global_load_async_to_lds_b128 %0, %1, off\n\t"
               "global_load_async_to_lds_b128 %0, %1, off offset:16"
               :
               : "v"(lds_addr), "v"(gaddr)
               : "memory");
}

__device__ __forceinline__ void wait_asynccnt0() {
  asm volatile("s_wait_asynccnt 0x0" ::: "memory");
}

// ------------------------------ tile staging -------------------------------

// A tile: BM x BK bf16 rows (optionally gathered) -> LDS via async DMA.
__device__ __forceinline__ void load_A_tile_async(__bf16* Al, const __bf16* A, int ldA,
                                                  int m0, int k0, const int* gather) {
  const int t   = threadIdx.x;       // 256 threads
  const int row = t >> 1;            // 0..127
  const int seg = (t & 1) << 4;      // 0 or 16 bf16 elements (32B per thread)
  const int grow = m0 + row;
  const int srow = gather ? gather[grow] : grow;
  const __bf16* src = A + (size_t)srow * ldA + k0 + seg;
  const unsigned lds = (unsigned)(uintptr_t)(Al + row * LDT + seg);
  async_cp_b128x2(lds, src);
}

// B tile: BN x BK, rows n0.. of an fp32 [N,K] row-major weight matrix,
// converted fp32 -> bf16 in registers, stored [n][k] into LDS.
__device__ __forceinline__ void load_B_tile(__bf16* Bl, const float* B, int ldB,
                                            int n0, int k0) {
  const int t   = threadIdx.x;
  const int row = t >> 1;
  const int seg = (t & 1) << 4;      // 0 or 16 elements
  const float4* s4 = (const float4*)(B + (size_t)(n0 + row) * ldB + k0 + seg);
  float4 a = s4[0], b = s4[1], c = s4[2], d = s4[3];
  union { __bf16 h[16]; u32x4 u[2]; } pk;
  pk.h[0]  = (__bf16)a.x; pk.h[1]  = (__bf16)a.y; pk.h[2]  = (__bf16)a.z; pk.h[3]  = (__bf16)a.w;
  pk.h[4]  = (__bf16)b.x; pk.h[5]  = (__bf16)b.y; pk.h[6]  = (__bf16)b.z; pk.h[7]  = (__bf16)b.w;
  pk.h[8]  = (__bf16)c.x; pk.h[9]  = (__bf16)c.y; pk.h[10] = (__bf16)c.z; pk.h[11] = (__bf16)c.w;
  pk.h[12] = (__bf16)d.x; pk.h[13] = (__bf16)d.y; pk.h[14] = (__bf16)d.z; pk.h[15] = (__bf16)d.w;
  u32x4* dst = (u32x4*)(Bl + row * LDT + seg);
  dst[0] = pk.u[0];
  dst[1] = pk.u[1];
}

// Fragment load per CDNA5 16-bit A/B striping:
// lane&15 selects the row (A) / col (B); lane>>4 selects the 8-wide K group,
// elements {kg..kg+7} and {16+kg..16+kg+7}  ->  two ds_load_b128.
__device__ __forceinline__ v16bf frag_ld(const __bf16* tile, int r0, int lane) {
  const int row = r0 + (lane & 15);
  const int kg  = (lane >> 4) << 3;   // 0 or 8
  const __bf16* p = tile + row * LDT + kg;
  union { v16bf v; u32x4 u[2]; } f;
  f.u[0] = *(const u32x4*)(p);
  f.u[1] = *(const u32x4*)(p + 16);
  return f.v;
}

__device__ __forceinline__ v8f wmma_bf16(v16bf a, v16bf b, v8f c) {
  return __builtin_amdgcn_wmma_f32_16x16x32_bf16(false, a, false, b, (short)0, c,
                                                 false, false);
}

// ------------------------------ gate / routing -----------------------------

__global__ __launch_bounds__(256)
void gate_topk_kernel(const float* __restrict__ x, const float* __restrict__ gw,
                      const float* __restrict__ gb, int* __restrict__ counts,
                      int* __restrict__ topk_id, float* __restrict__ topk_w,
                      int* __restrict__ pos_out) {
  const int wave = threadIdx.x >> 5;
  const int lane = threadIdx.x & 31;
  const int t = blockIdx.x * 8 + wave;

  float acc[E_NUM];
#pragma unroll
  for (int e = 0; e < E_NUM; ++e) acc[e] = 0.f;

  const float* xr = x + (size_t)t * H_DIM;
  for (int h = lane; h < H_DIM; h += 32) {
    float xv = xr[h];
#pragma unroll
    for (int e = 0; e < E_NUM; ++e) acc[e] += xv * gw[e * H_DIM + h];
  }
#pragma unroll
  for (int e = 0; e < E_NUM; ++e)
    for (int m = 16; m >= 1; m >>= 1) acc[e] += __shfl_xor(acc[e], m, 32);

  if (lane != 0) return;

  float s[E_NUM], sc[E_NUM];
#pragma unroll
  for (int e = 0; e < E_NUM; ++e) {
    s[e]  = 1.f / (1.f + __expf(-acc[e]));
    sc[e] = s[e] + gb[e];
  }
  // per-group top-2 sum (4 groups of 4)
  float gsum[4];
#pragma unroll
  for (int g = 0; g < 4; ++g) {
    float m1 = -1e30f, m2 = -1e30f;
#pragma unroll
    for (int i = 0; i < 4; ++i) {
      float v = sc[g * 4 + i];
      if (v > m1) { m2 = m1; m1 = v; } else if (v > m2) { m2 = v; }
    }
    gsum[g] = m1 + m2;
  }
  int g1 = 0;
  for (int g = 1; g < 4; ++g) if (gsum[g] > gsum[g1]) g1 = g;
  int g2 = -1;
  for (int g = 0; g < 4; ++g)
    if (g != g1 && (g2 < 0 || gsum[g] > gsum[g2])) g2 = g;

  bool used[E_NUM];
#pragma unroll
  for (int e = 0; e < E_NUM; ++e) used[e] = false;

  int ids[TOP_K];
  float wv[TOP_K];
  float wsum = 0.f;
#pragma unroll
  for (int j = 0; j < TOP_K; ++j) {
    int best = -1;
    float bv = -1e30f;
    for (int e = 0; e < E_NUM; ++e) {
      int g = e >> 2;
      if ((g == g1 || g == g2) && !used[e] && sc[e] > bv) { bv = sc[e]; best = e; }
    }
    used[best] = true;
    ids[j] = best;
    wv[j] = s[best];
    wsum += s[best];
  }
  const float scale = ROUTED_SCALE / wsum;
#pragma unroll
  for (int j = 0; j < TOP_K; ++j) {
    topk_id[t * TOP_K + j] = ids[j];
    topk_w[t * TOP_K + j] = wv[j] * scale;
    pos_out[t * TOP_K + j] = atomicAdd(&counts[ids[j]], 1);
  }
}

__global__ void scan_kernel(const int* __restrict__ counts, int* __restrict__ offsets,
                            int* __restrict__ tile_expert, int* __restrict__ row_token) {
  if (threadIdx.x == 0) {
    int offl[E_NUM + 1];
    int off = 0;
    for (int e = 0; e < E_NUM; ++e) {
      offsets[e] = off;
      offl[e] = off;
      off += ((counts[e] + BM - 1) / BM) * BM;
    }
    offl[E_NUM] = off;
    for (int tl = 0; tl < ROW_TILES; ++tl) {
      int rb = tl * BM;
      int e = -1;
      for (int j = 0; j < E_NUM; ++j)
        if (rb >= offl[j] && rb < offl[j + 1]) e = j;
      tile_expert[tl] = e;
    }
  }
  for (int i = threadIdx.x; i < CAP_ROWS; i += blockDim.x) row_token[i] = 0;
}

__global__ void build_rows_kernel(const int* __restrict__ topk_id,
                                  const int* __restrict__ pos,
                                  const int* __restrict__ offsets,
                                  int* __restrict__ row_token) {
  int t = blockIdx.x * blockDim.x + threadIdx.x;
  if (t >= T_TOK) return;
#pragma unroll
  for (int j = 0; j < TOP_K; ++j) {
    int e = topk_id[t * TOP_K + j];
    int r = offsets[e] + pos[t * TOP_K + j];
    row_token[r] = t;
  }
}

__global__ void cvt_bf16_kernel(const float* __restrict__ x, __bf16* __restrict__ y) {
  int i = (blockIdx.x * blockDim.x + threadIdx.x) * 8;
  const float4* s = (const float4*)(x + i);
  float4 a = s[0], b = s[1];
  union { __bf16 h[8]; u32x4 u; } pk;
  pk.h[0] = (__bf16)a.x; pk.h[1] = (__bf16)a.y; pk.h[2] = (__bf16)a.z; pk.h[3] = (__bf16)a.w;
  pk.h[4] = (__bf16)b.x; pk.h[5] = (__bf16)b.y; pk.h[6] = (__bf16)b.z; pk.h[7] = (__bf16)b.w;
  *(u32x4*)(y + i) = pk.u;
}

// ----------------------- up-projection + fused SwiGLU ----------------------
// C_g = A @ Bg^T, C_u = A @ Bu^T  (Bg = W rows [0,Nh), Bu = rows [Nh,2Nh))
// act = silu(C_g) * C_u -> bf16.  8 waves, each wave owns a 64x32 subtile.
// Double-buffered LDS: stage tile k+1 (A via async DMA, B via load+cvt+ds)
// while WMMAs consume tile k; single barrier per K-step.
template <bool ROUTED>
__global__ __launch_bounds__(256)
void swiglu_gemm_kernel(const __bf16* __restrict__ A, const float* __restrict__ B,
                        __bf16* __restrict__ act, int K, int Nh, int ldA,
                        const int* __restrict__ row_token,
                        const int* __restrict__ tile_expert,
                        const int* __restrict__ offsets,
                        const int* __restrict__ counts, long wstride) {
  const int m0 = blockIdx.y * BM;
  const int n0 = blockIdx.x * BN;
  const float* Bp = B;
  const int* gather = nullptr;
  if constexpr (ROUTED) {
    int e = tile_expert[blockIdx.y];
    if (e < 0) return;
    if (m0 >= offsets[e] + counts[e]) return;   // fully-padding tile
    Bp = B + (size_t)e * wstride;
    gather = row_token;
  }

  __shared__ __attribute__((aligned(16))) __bf16 Al[2][BM * LDT];
  __shared__ __attribute__((aligned(16))) __bf16 Bg[2][BN * LDT];
  __shared__ __attribute__((aligned(16))) __bf16 Bu[2][BN * LDT];

  const int lane = threadIdx.x & 31;
  const int wave = threadIdx.x >> 5;
  const int wm = wave & 1;   // 2 wave-rows  (64 rows each)
  const int wn = wave >> 1;  // 4 wave-cols  (32 cols each)

  v8f accg[4][2], accu[4][2];
  const v8f vzero = {0.f, 0.f, 0.f, 0.f, 0.f, 0.f, 0.f, 0.f};
#pragma unroll
  for (int im = 0; im < 4; ++im)
#pragma unroll
    for (int in_ = 0; in_ < 2; ++in_) { accg[im][in_] = vzero; accu[im][in_] = vzero; }

  // prologue: stage tile 0 into buffer 0
  load_A_tile_async(Al[0], A, ldA, m0, 0, gather);
  load_B_tile(Bg[0], Bp, K, n0, 0);
  load_B_tile(Bu[0], Bp, K, n0 + Nh, 0);
  wait_asynccnt0();
  __syncthreads();

  int cur = 0;
  for (int k0 = 0; k0 < K; k0 += BK) {
    const int nxt = cur ^ 1;
    if (k0 + BK < K) {
      load_A_tile_async(Al[nxt], A, ldA, m0, k0 + BK, gather);
      load_B_tile(Bg[nxt], Bp, K, n0, k0 + BK);
      load_B_tile(Bu[nxt], Bp, K, n0 + Nh, k0 + BK);
      __builtin_prefetch(Bp + (size_t)(n0 + (threadIdx.x >> 1)) * K + k0 + 2 * BK, 0, 1);
    }

    v16bf af[4];
#pragma unroll
    for (int im = 0; im < 4; ++im) af[im] = frag_ld(Al[cur], wm * 64 + im * 16, lane);
#pragma unroll
    for (int in_ = 0; in_ < 2; ++in_) {
      v16bf bg = frag_ld(Bg[cur], wn * 32 + in_ * 16, lane);
      v16bf bu = frag_ld(Bu[cur], wn * 32 + in_ * 16, lane);
#pragma unroll
      for (int im = 0; im < 4; ++im) {
        accg[im][in_] = wmma_bf16(af[im], bg, accg[im][in_]);
        accu[im][in_] = wmma_bf16(af[im], bu, accu[im][in_]);
      }
    }

    wait_asynccnt0();
    __syncthreads();
    cur = nxt;
  }

  // epilogue: act = silu(g)*u -> bf16, row-major [*, Nh]
#pragma unroll
  for (int im = 0; im < 4; ++im) {
#pragma unroll
    for (int in_ = 0; in_ < 2; ++in_) {
      const int mbase = m0 + wm * 64 + im * 16 + ((lane < 16) ? 0 : 8);
      const int nc    = n0 + wn * 32 + in_ * 16 + (lane & 15);
#pragma unroll
      for (int r = 0; r < 8; ++r) {
        float g = accg[im][in_][r];
        float u = accu[im][in_][r];
        float a = (g / (1.f + __expf(-g))) * u;
        act[(size_t)(mbase + r) * Nh + nc] = (__bf16)a;
      }
    }
  }
}

// ------------------------------ down-projection ----------------------------
// C[M, H] = A_bf16[M, K] @ B_fp32[H, K]^T ; fp32 output (shared -> d_out,
// routed -> compact y buffer, combined later with routing weights).
template <bool ROUTED>
__global__ __launch_bounds__(256)
void down_gemm_kernel(const __bf16* __restrict__ A, const float* __restrict__ B,
                      float* __restrict__ Cout, int K,
                      const int* __restrict__ tile_expert,
                      const int* __restrict__ offsets,
                      const int* __restrict__ counts, long wstride) {
  const int m0 = blockIdx.y * BM;
  const int n0 = blockIdx.x * BN;
  const float* Bp = B;
  if constexpr (ROUTED) {
    int e = tile_expert[blockIdx.y];
    if (e < 0) return;
    if (m0 >= offsets[e] + counts[e]) return;
    Bp = B + (size_t)e * wstride;
  }

  __shared__ __attribute__((aligned(16))) __bf16 Al[2][BM * LDT];
  __shared__ __attribute__((aligned(16))) __bf16 Bl[2][BN * LDT];

  const int lane = threadIdx.x & 31;
  const int wave = threadIdx.x >> 5;
  const int wm = wave & 1;
  const int wn = wave >> 1;

  v8f acc[4][2];
  const v8f vzero = {0.f, 0.f, 0.f, 0.f, 0.f, 0.f, 0.f, 0.f};
#pragma unroll
  for (int im = 0; im < 4; ++im)
#pragma unroll
    for (int in_ = 0; in_ < 2; ++in_) acc[im][in_] = vzero;

  load_A_tile_async(Al[0], A, K, m0, 0, nullptr);
  load_B_tile(Bl[0], Bp, K, n0, 0);
  wait_asynccnt0();
  __syncthreads();

  int cur = 0;
  for (int k0 = 0; k0 < K; k0 += BK) {
    const int nxt = cur ^ 1;
    if (k0 + BK < K) {
      load_A_tile_async(Al[nxt], A, K, m0, k0 + BK, nullptr);
      load_B_tile(Bl[nxt], Bp, K, n0, k0 + BK);
      __builtin_prefetch(Bp + (size_t)(n0 + (threadIdx.x >> 1)) * K + k0 + 2 * BK, 0, 1);
    }

    v16bf af[4];
#pragma unroll
    for (int im = 0; im < 4; ++im) af[im] = frag_ld(Al[cur], wm * 64 + im * 16, lane);
#pragma unroll
    for (int in_ = 0; in_ < 2; ++in_) {
      v16bf bf = frag_ld(Bl[cur], wn * 32 + in_ * 16, lane);
#pragma unroll
      for (int im = 0; im < 4; ++im) acc[im][in_] = wmma_bf16(af[im], bf, acc[im][in_]);
    }

    wait_asynccnt0();
    __syncthreads();
    cur = nxt;
  }

#pragma unroll
  for (int im = 0; im < 4; ++im) {
#pragma unroll
    for (int in_ = 0; in_ < 2; ++in_) {
      const int mbase = m0 + wm * 64 + im * 16 + ((lane < 16) ? 0 : 8);
      const int nc    = n0 + wn * 32 + in_ * 16 + (lane & 15);
#pragma unroll
      for (int r = 0; r < 8; ++r)
        Cout[(size_t)(mbase + r) * H_DIM + nc] = acc[im][in_][r];
    }
  }
}

// ------------------------------- combine -----------------------------------
// out[t] += sum_j w[t][j] * y_routed[row(t,j)]   (deterministic, no atomics)
__global__ __launch_bounds__(256)
void combine_kernel(float* __restrict__ out, const float* __restrict__ y,
                    const int* __restrict__ topk_id, const float* __restrict__ topk_w,
                    const int* __restrict__ pos, const int* __restrict__ offsets) {
  const int t = blockIdx.x;
  int rows[TOP_K];
  float w[TOP_K];
#pragma unroll
  for (int j = 0; j < TOP_K; ++j) {
    int e = topk_id[t * TOP_K + j];
    rows[j] = offsets[e] + pos[t * TOP_K + j];
    w[j] = topk_w[t * TOP_K + j];
  }
  for (int h = threadIdx.x; h < H_DIM; h += blockDim.x) {
    float a = out[(size_t)t * H_DIM + h];
#pragma unroll
    for (int j = 0; j < TOP_K; ++j) a += w[j] * y[(size_t)rows[j] * H_DIM + h];
    out[(size_t)t * H_DIM + h] = a;
  }
}

// ------------------------------- launcher ----------------------------------

extern "C" void kernel_launch(void* const* d_in, const int* in_sizes, int n_in,
                              void* d_out, int out_size, void* d_ws, size_t ws_size,
                              hipStream_t stream) {
  (void)in_sizes; (void)n_in; (void)out_size; (void)ws_size;
  const float* x   = (const float*)d_in[0];   // [T, H]
  const float* gw  = (const float*)d_in[1];   // [E, H]
  const float* gb  = (const float*)d_in[2];   // [E]
  const float* w13 = (const float*)d_in[3];   // [E, 2I, H]
  const float* w2  = (const float*)d_in[4];   // [E, H, I]
  const float* sgu = (const float*)d_in[5];   // [2*IS, H]
  const float* sdn = (const float*)d_in[6];   // [H, IS]
  float* out = (float*)d_out;                 // [T, H]

  char* p = (char*)d_ws;
  auto carve = [&](size_t bytes) -> char* {
    char* r = p;
    p += (bytes + 255) & ~(size_t)255;
    return r;
  };
  int*    counts      = (int*)carve(E_NUM * sizeof(int));
  int*    offsets     = (int*)carve(E_NUM * sizeof(int));
  int*    tile_expert = (int*)carve(ROW_TILES * sizeof(int));
  int*    topk_id     = (int*)carve((size_t)T_TOK * TOP_K * sizeof(int));
  float*  topk_w      = (float*)carve((size_t)T_TOK * TOP_K * sizeof(float));
  int*    pos         = (int*)carve((size_t)T_TOK * TOP_K * sizeof(int));
  int*    row_token   = (int*)carve((size_t)CAP_ROWS * sizeof(int));
  __bf16* x_bf        = (__bf16*)carve((size_t)T_TOK * H_DIM * sizeof(__bf16));
  __bf16* act_shared  = (__bf16*)carve((size_t)T_TOK * IS_DIM * sizeof(__bf16));
  __bf16* act_routed  = (__bf16*)carve((size_t)CAP_ROWS * I_DIM * sizeof(__bf16));
  float*  y_routed    = (float*)carve((size_t)CAP_ROWS * H_DIM * sizeof(float));

  hipMemsetAsync(counts, 0, E_NUM * sizeof(int), stream);

  // x -> bf16
  cvt_bf16_kernel<<<(T_TOK * H_DIM) / (256 * 8), 256, 0, stream>>>(x, x_bf);

  // routing
  gate_topk_kernel<<<T_TOK / 8, 256, 0, stream>>>(x, gw, gb, counts, topk_id, topk_w, pos);
  scan_kernel<<<1, 256, 0, stream>>>(counts, offsets, tile_expert, row_token);
  build_rows_kernel<<<(T_TOK + 255) / 256, 256, 0, stream>>>(topk_id, pos, offsets, row_token);

  // shared expert: gate_up + SwiGLU
  swiglu_gemm_kernel<false><<<dim3(IS_DIM / BN, T_TOK / BM), 256, 0, stream>>>(
      x_bf, sgu, act_shared, H_DIM, IS_DIM, H_DIM, nullptr, nullptr, nullptr, nullptr, 0);

  // routed experts: gathered gate_up + SwiGLU into compact rows
  swiglu_gemm_kernel<true><<<dim3(I_DIM / BN, ROW_TILES), 256, 0, stream>>>(
      x_bf, w13, act_routed, H_DIM, I_DIM, H_DIM, row_token, tile_expert, offsets, counts,
      (long)(2 * I_DIM) * H_DIM);

  // shared down-proj -> out (each element written exactly once)
  down_gemm_kernel<false><<<dim3(H_DIM / BN, T_TOK / BM), 256, 0, stream>>>(
      act_shared, sdn, out, IS_DIM, nullptr, nullptr, nullptr, 0);

  // routed down-proj -> compact y buffer
  down_gemm_kernel<true><<<dim3(H_DIM / BN, ROW_TILES), 256, 0, stream>>>(
      act_routed, w2, y_routed, I_DIM, tile_expert, offsets, counts, (long)H_DIM * I_DIM);

  // out += weighted routed outputs
  combine_kernel<<<T_TOK, 256, 0, stream>>>(out, y_routed, topk_id, topk_w, pos, offsets);
}
